// GatedLinearAttention_39015482916949
// MI455X (gfx1250) — matrix operand order repaired and checked by reading, not compiled, not run on hardware
//
#include <hip/hip_runtime.h>
#include <hip/hip_bf16.h>

typedef __attribute__((ext_vector_type(16))) _Float16 v16h;
typedef __attribute__((ext_vector_type(8)))  _Float16 v8h;
typedef __attribute__((ext_vector_type(8)))  float    v8f;

#define B_  2
#define L_  2048
#define D_  2048
#define H_  16
#define DK_ 64
#define DV_ 128
#define M_  (B_ * L_)   // 4096 rows

// ---------------------------------------------------------------------------
// fp32 -> f16 conversion (grid-stride)
// ---------------------------------------------------------------------------
__global__ void cvt_f32_to_f16(const float* __restrict__ in,
                               _Float16* __restrict__ out, size_t n) {
  size_t i = (size_t)blockIdx.x * blockDim.x + threadIdx.x;
  size_t stride = (size_t)gridDim.x * blockDim.x;
  for (; i < n; i += stride) out[i] = (_Float16)in[i];
}

// ---------------------------------------------------------------------------
// WMMA GEMM: C[M,N] = scale * A[M,K] @ W[N,K]^T   (A,W f16 row-major; C f32)
// wave tile = 64(M) x 64(N), K stepped by 32: 16 WMMAs per 16 b128 loads.
// Fragment layouts: A/B use k = 16*half + s (same order in both); f32 C/D per CDNA5 ISA 7.12.2.
// ---------------------------------------------------------------------------
__global__ __launch_bounds__(256)
void gemm_wmma_f16(const _Float16* __restrict__ A, const _Float16* __restrict__ W,
                   float* __restrict__ C, int M, int N, int K, float scale) {
  const int wave = threadIdx.x >> 5;
  const int lane = threadIdx.x & 31;
  const int half = lane >> 4;   // 0: lanes 0-15, 1: lanes 16-31
  const int l16  = lane & 15;

  const int tiles_n = N >> 6;
  const int wt = blockIdx.x * 8 + wave;
  const int tm = wt / tiles_n;
  const int tn = wt % tiles_n;
  if (tm * 64 >= M) return;
  const int m0 = tm * 64;
  const int n0 = tn * 64;

  v8f acc[4][4];
#pragma unroll
  for (int i = 0; i < 4; i++)
#pragma unroll
    for (int j = 0; j < 4; j++)
#pragma unroll
      for (int r = 0; r < 8; r++) acc[i][j][r] = 0.0f;

  const _Float16* Ar[4];
  const _Float16* Wr[4];
#pragma unroll
  for (int i = 0; i < 4; i++) Ar[i] = A + (size_t)(m0 + i * 16 + l16) * K;
#pragma unroll
  for (int j = 0; j < 4; j++) Wr[j] = W + (size_t)(n0 + j * 16 + l16) * K;
  const int kA = half * 16;   // A lane K base within 32-chunk
  const int kB = half * 16;   // B lane K base within 32-chunk

  for (int k0 = 0; k0 < K; k0 += 32) {
    v16h a[4], b[4];
#pragma unroll
    for (int i = 0; i < 4; i++) {
      // lane holds K {kA..kA+15} of its row
      a[i] = *(const v16h*)(Ar[i] + k0 + kA);
    }
#pragma unroll
    for (int j = 0; j < 4; j++)
      // B column N = n-row of W: contiguous 16 halves starting at k0+kB
      b[j] = *(const v16h*)(Wr[j] + k0 + kB);

#pragma unroll
    for (int i = 0; i < 4; i++)
#pragma unroll
      for (int j = 0; j < 4; j++)
        acc[i][j] = __builtin_amdgcn_wmma_f32_16x16x32_f16(
            false, a[i], false, b[j], (short)0, acc[i][j], false, false);
  }

  // C/D layout: VGPR r, lanes 0-15 -> M=r, N=lane ; lanes 16-31 -> M=r+8
#pragma unroll
  for (int i = 0; i < 4; i++)
#pragma unroll
    for (int j = 0; j < 4; j++)
#pragma unroll
      for (int r = 0; r < 8; r++) {
        int row = m0 + i * 16 + half * 8 + r;
        int col = n0 + j * 16 + l16;
        C[(size_t)row * N + col] = acc[i][j][r] * scale;
      }
}

// ---------------------------------------------------------------------------
// kg stage 1: y1[M,16] = x[M,2048] @ Wkg1[16,2048]^T   (fp32, tiny)
// one block per row; thread t: j = t&15 output, c = t>>4 K-slice of 128
// ---------------------------------------------------------------------------
__global__ __launch_bounds__(256)
void kg1_kernel(const float* __restrict__ x, const float* __restrict__ Wkg1,
                float* __restrict__ y1) {
  const int row = blockIdx.x;
  const int t = threadIdx.x;
  const int j = t & 15;
  const int c = t >> 4;
  const float* xr = x + (size_t)row * D_ + c * 128;
  const float* wr = Wkg1 + (size_t)j * D_ + c * 128;
  float acc = 0.f;
#pragma unroll 8
  for (int k = 0; k < 128; k++) acc = fmaf(xr[k], wr[k], acc);
  __shared__ float red[256];
  red[t] = acc;
  __syncthreads();
  if (t < 16) {
    float s = 0.f;
#pragma unroll
    for (int cc = 0; cc < 16; cc++) s += red[t + (cc << 4)];
    y1[(size_t)row * 16 + t] = s;
  }
}

// ---------------------------------------------------------------------------
// kg stage 2 + gate: eg = exp( logsigmoid(y1 @ Wkg2^T + bkg2) / 16 )
// ---------------------------------------------------------------------------
__global__ __launch_bounds__(256)
void kg2_kernel(const float* __restrict__ y1, const float* __restrict__ Wkg2,
                const float* __restrict__ bkg2, float* __restrict__ eg) {
  const int i = blockIdx.x * blockDim.x + threadIdx.x;   // [0, M*1024)
  const int n = i & 1023;
  const int row = i >> 10;
  const float* y = y1 + (size_t)row * 16;
  const float* w = Wkg2 + (size_t)n * 16;
  float acc = bkg2[n];
#pragma unroll
  for (int j = 0; j < 16; j++) acc = fmaf(y[j], w[j], acc);
  // stable log-sigmoid
  float ls = fminf(acc, 0.f) - __logf(1.f + __expf(-fabsf(acc)));
  eg[i] = __expf(ls * (1.0f / 16.0f));
}

// ---------------------------------------------------------------------------
// GLA recurrence: one block per (b,h), thread = DV column, state in VGPRs.
// S[dk,dv] = eg[dk]*S + k[dk]*v[dv] ; o[dv] = sum_dk q[dk]*S[dk,dv]
// Software-pipelined: q/k/eg for step t+1 load into registers during the
// FMA chain of step t; next v row warmed via global_prefetch.
// o overwrites v in place (same per-thread address, read-before-write).
// ---------------------------------------------------------------------------
__global__ __launch_bounds__(128)
void gla_recurrence(const float* __restrict__ q, const float* __restrict__ k,
                    const float* __restrict__ eg, float* __restrict__ v,
                    const float* __restrict__ hs, float* __restrict__ s_out) {
  const int bh = blockIdx.x;        // 0..31
  const int b = bh >> 4;
  const int h = bh & 15;
  const int dv = threadIdx.x;       // 0..127

  float S[DK_];
  const float* hsp = hs + (size_t)bh * DK_ * DV_ + dv;
#pragma unroll
  for (int d = 0; d < DK_; d++) S[d] = hsp[(size_t)d * DV_];

  __shared__ float qs[DK_], ks[DK_], es[DK_];

  const float* qb = q + (size_t)b * L_ * 1024 + h * DK_;
  const float* kb = k + (size_t)b * L_ * 1024 + h * DK_;
  const float* eb = eg + (size_t)b * L_ * 1024 + h * DK_;
  float* vb = v + (size_t)b * L_ * D_ + h * DV_;

  // preload step 0 into registers (no wave divergence: dv<64 == waves 0,1)
  float rq = 0.f, re = 0.f, rk = 0.f;
  if (dv < 64) { rq = qb[dv]; re = eb[dv]; }
  else         { rk = kb[dv - 64]; }

  for (int t = 0; t < L_; t++) {
    if (dv < 64) { qs[dv] = rq; es[dv] = re; }
    else         { ks[dv - 64] = rk; }
    __syncthreads();

    // issue next step's loads; they complete under this step's FMA chain
    if (t + 1 < L_) {
      const size_t o1 = (size_t)(t + 1) * 1024;
      if (dv < 64) { rq = qb[o1 + dv]; re = eb[o1 + dv]; }
      else         { rk = kb[o1 + dv - 64]; }
      __builtin_prefetch(vb + (size_t)(t + 1) * D_ + dv, 0, 3);
    }

    float vv = vb[(size_t)t * D_ + dv];
    float acc = 0.f;
#pragma unroll
    for (int d = 0; d < DK_; d++) {
      S[d] = fmaf(es[d], S[d], ks[d] * vv);
      acc = fmaf(qs[d], S[d], acc);
    }
    vb[(size_t)t * D_ + dv] = acc;   // o_t replaces v_t
    __syncthreads();
  }

  float* sp = s_out + (size_t)bh * DK_ * DV_ + dv;
#pragma unroll
  for (int d = 0; d < DK_; d++) sp[(size_t)d * DV_] = S[d];
}

// ---------------------------------------------------------------------------
// LayerNorm(head dim, no affine) + silu(g + bg) multiply -> f16 y
// one block per (b,l,h), 128 threads
// ---------------------------------------------------------------------------
__global__ __launch_bounds__(128)
void ln_silu_kernel(const float* __restrict__ o, const float* __restrict__ g,
                    const float* __restrict__ bg, _Float16* __restrict__ y) {
  const int idx = blockIdx.x;       // (b*L + l)*H + h
  const int h = idx & 15;
  const int bl = idx >> 4;
  const int dv = threadIdx.x;
  const size_t off = (size_t)bl * D_ + h * DV_ + dv;

  float val = o[off];
  float s = val, s2 = val * val;
#pragma unroll
  for (int m = 16; m >= 1; m >>= 1) {
    s += __shfl_xor(s, m, 32);
    s2 += __shfl_xor(s2, m, 32);
  }
  __shared__ float ps[4], ps2[4];
  const int w = threadIdx.x >> 5, lane = threadIdx.x & 31;
  if (lane == 0) { ps[w] = s; ps2[w] = s2; }
  __syncthreads();
  float tot = ps[0] + ps[1] + ps[2] + ps[3];
  float tot2 = ps2[0] + ps2[1] + ps2[2] + ps2[3];
  float mu = tot * (1.0f / 128.0f);
  float var = tot2 * (1.0f / 128.0f) - mu * mu;
  float nrm = (val - mu) * __frsqrt_rn(var + 1e-5f);

  float gv = g[off] + bg[h * DV_ + dv];
  float sil = gv / (1.0f + __expf(-gv));
  y[off] = (_Float16)(sil * nrm);
}

// ---------------------------------------------------------------------------
// launch
// ---------------------------------------------------------------------------
extern "C" void kernel_launch(void* const* d_in, const int* in_sizes, int n_in,
                              void* d_out, int out_size, void* d_ws, size_t ws_size,
                              hipStream_t stream) {
  const float* x    = (const float*)d_in[0];
  const float* hs   = (const float*)d_in[1];
  const float* Wq   = (const float*)d_in[2];
  const float* Wk   = (const float*)d_in[3];
  const float* Wkg1 = (const float*)d_in[4];
  const float* Wkg2 = (const float*)d_in[5];
  const float* bkg2 = (const float*)d_in[6];
  const float* Wv   = (const float*)d_in[7];
  const float* Wg   = (const float*)d_in[8];
  const float* bg   = (const float*)d_in[9];
  const float* Wo   = (const float*)d_in[10];

  float* out = (float*)d_out;                       // [B,L,D]
  float* s_final = out + (size_t)M_ * D_;           // [B,H,DK,DV]

  // workspace carve-up (256B aligned)
  char* ws = (char*)d_ws;
  size_t off = 0;
  auto take = [&](size_t bytes) { char* p = ws + off; off += (bytes + 255) & ~(size_t)255; return p; };
  _Float16* x_h  = (_Float16*)take((size_t)M_ * D_ * 2);        // reused as y_h later
  _Float16* wq_h = (_Float16*)take((size_t)1024 * D_ * 2);
  _Float16* wk_h = (_Float16*)take((size_t)1024 * D_ * 2);
  _Float16* wv_h = (_Float16*)take((size_t)D_ * D_ * 2);
  _Float16* wg_h = (_Float16*)take((size_t)D_ * D_ * 2);
  _Float16* wo_h = (_Float16*)take((size_t)D_ * D_ * 2);
  float* qbuf = (float*)take((size_t)M_ * 1024 * 4);
  float* kbuf = (float*)take((size_t)M_ * 1024 * 4);
  float* egbuf = (float*)take((size_t)M_ * 1024 * 4);
  float* vbuf = (float*)take((size_t)M_ * D_ * 4);              // v, then o in place
  float* gbuf = (float*)take((size_t)M_ * D_ * 4);
  float* y1buf = (float*)take((size_t)M_ * 16 * 4);
  (void)off; (void)ws_size; (void)in_sizes; (void)n_in; (void)out_size;

  // 1) f32 -> f16 conversions
  cvt_f32_to_f16<<<2048, 256, 0, stream>>>(x, x_h, (size_t)M_ * D_);
  cvt_f32_to_f16<<<1024, 256, 0, stream>>>(Wq, wq_h, (size_t)1024 * D_);
  cvt_f32_to_f16<<<1024, 256, 0, stream>>>(Wk, wk_h, (size_t)1024 * D_);
  cvt_f32_to_f16<<<2048, 256, 0, stream>>>(Wv, wv_h, (size_t)D_ * D_);
  cvt_f32_to_f16<<<2048, 256, 0, stream>>>(Wg, wg_h, (size_t)D_ * D_);
  cvt_f32_to_f16<<<2048, 256, 0, stream>>>(Wo, wo_h, (size_t)D_ * D_);

  // 2) projection GEMMs (WMMA); grid = (M/64 * N/64) / 8 waves-per-block
  const float kscale = 0.08838834764831845f;  // (D/H)^-0.5 = 128^-0.5
  gemm_wmma_f16<<<128, 256, 0, stream>>>(x_h, wq_h, qbuf, M_, 1024, D_, 1.0f);
  gemm_wmma_f16<<<128, 256, 0, stream>>>(x_h, wk_h, kbuf, M_, 1024, D_, kscale);
  gemm_wmma_f16<<<256, 256, 0, stream>>>(x_h, wv_h, vbuf, M_, D_, D_, 1.0f);
  gemm_wmma_f16<<<256, 256, 0, stream>>>(x_h, wg_h, gbuf, M_, D_, D_, 1.0f);

  // 3) low-rank decay gate -> eg = exp(logsigmoid(kg)/16)
  kg1_kernel<<<M_, 256, 0, stream>>>(x, Wkg1, y1buf);
  kg2_kernel<<<(M_ * 1024) / 256, 256, 0, stream>>>(y1buf, Wkg2, bkg2, egbuf);

  // 4) serial GLA recurrence; o overwrites v; S_final to output tail
  gla_recurrence<<<B_ * H_, 128, 0, stream>>>(qbuf, kbuf, egbuf, vbuf, hs, s_final);

  // 5) per-head LayerNorm + silu(g) gate -> f16 y (reuses x_h buffer)
  ln_silu_kernel<<<M_ * H_, 128, 0, stream>>>(vbuf, gbuf, bg, x_h);

  // 6) output projection (WMMA) -> d_out
  gemm_wmma_f16<<<256, 256, 0, stream>>>(x_h, wo_h, out, M_, D_, D_, 1.0f);
}
